// Transformer2Layer_32899449487820
// MI455X (gfx1250) — compile-verified, matrix operand order checked by reading, not run on hardware
//
#include <hip/hip_runtime.h>
#include <hip/hip_bf16.h>

typedef __attribute__((ext_vector_type(16))) _Float16 v16h;
typedef __attribute__((ext_vector_type(8)))  float    v8f;

#define SEQ_LEN 1024
#define BATCH   128
#define MLP_H   512

// ---------------------------------------------------------------------------
// Kernel 1: stage-1 attention for all (s, b).
//   y[s,b] = sum_t softmax_t( q[s,b]*k[t,b] + bias[s,t] ) * v[t,b]
// q/k/v are scalar affine maps of src (HIDDEN == 1).
// Block: 256 threads (8 waves). blockIdx.x -> s-tile (16 rows, shared by all
// waves), blockIdx.y -> group of 8 b's (one b per wave).
// Score tile computed with V_WMMA_F32_16X16X32_F16: A = q column in K=0,
// B = k row in K=0, C = bias tile (bias add folded into the WMMA).
// Softmax without max-subtraction (shift-invariant; scores O(1) bounded).
// ---------------------------------------------------------------------------
__global__ __launch_bounds__(256)
void attn1_kernel(const float* __restrict__ src,
                  const float* __restrict__ Wq, const float* __restrict__ bq,
                  const float* __restrict__ Wk, const float* __restrict__ bk,
                  const float* __restrict__ Wv, const float* __restrict__ bv,
                  const float* __restrict__ bias,
                  float* __restrict__ y)
{
    __shared__ float cols[8][SEQ_LEN + 4];   // 8 src columns (one per wave's b)

    const int tid  = threadIdx.x;
    const int lane = tid & 31;
    const int wv   = tid >> 5;               // wave id = which b in the group
    const int s0   = blockIdx.x * 16;        // s-tile start (shared by waves)
    const int b0   = blockIdx.y * 8;

    // Cooperative load of the 8 src columns: cols[bi][t] = src[t, b0+bi]
    for (int idx = tid; idx < 8 * SEQ_LEN; idx += 256) {
        int bi = idx & 7;
        int t  = idx >> 3;
        cols[bi][t] = src[(size_t)t * BATCH + b0 + bi];
    }
    __syncthreads();

    const float wqv = Wq[0], bqv = bq[0];
    const float wkv = Wk[0], bkv = bk[0];
    const float wvv = Wv[0], bvv = bv[0];

    // A operand: q column in K=0.  16-bit A 16x32 layout: VGPR0 lanes0-15
    // hold K=0,1 for M=lane; lanes16-31 hold K=8,9 (must stay zero).
    v16h amat = {};
    if (lane < 16) {
        float qv = fmaf(cols[wv][s0 + lane], wqv, bqv);
        amat[0] = (_Float16)qv;
    }

    float l_part[8];
    float a_part[8];
#pragma unroll
    for (int m = 0; m < 8; ++m) { l_part[m] = 0.f; a_part[m] = 0.f; }

    const int rowbase = (lane >> 4) << 3;    // 0 for lanes 0-15, 8 for 16-31
    const int tl15 = lane & 15;

    for (int t0 = 0; t0 < SEQ_LEN; t0 += 16) {
        const int tl = t0 + tl15;
        const float colv = cols[wv][tl];             // src[tl, b]

        // B operand: k row in K=0.  16-bit B 32x16 layout: VGPR0 lanes0-15
        // hold K=0,1 for N=lane; lanes16-31 hold K=8,9 (must stay zero).
        v16h bmat = {};
        if (lane < 16) bmat[0] = (_Float16)fmaf(colv, wkv, bkv);

        const float vt = fmaf(colv, wvv, bvv);       // v[tl, b] (all lanes)

        // C = bias tile: lane -> column t'=tl, VGPR m -> row s0+rowbase+m
        const float* bp = bias + (size_t)(s0 + rowbase) * SEQ_LEN + tl;
        v8f cv;
#pragma unroll
        for (int m = 0; m < 8; ++m) cv[m] = bp[(size_t)m * SEQ_LEN];

        if (t0 + 16 < SEQ_LEN)
            __builtin_prefetch(bp + 16, 0, 1);       // next bias tile

        // scores[s, t'] = q[s]*k[t'] + bias[s, t']
        v8f d = __builtin_amdgcn_wmma_f32_16x16x32_f16(
            /*neg_a=*/false, amat, /*neg_b=*/false, bmat,
            /*c_mod=*/(short)0, cv, /*reuse_a=*/false, /*reuse_b=*/false);

#pragma unroll
        for (int m = 0; m < 8; ++m) {
            float p = __expf(d[m]);
            l_part[m] += p;
            a_part[m] = fmaf(p, vt, a_part[m]);
        }
    }

    // Reduce across the 16 lanes holding each row (xor masks 1..8 stay
    // within each 16-lane half of the wave32).
#pragma unroll
    for (int m = 0; m < 8; ++m) {
        float lv = l_part[m], av = a_part[m];
#pragma unroll
        for (int mask = 8; mask >= 1; mask >>= 1) {
            lv += __shfl_xor(lv, mask, 32);
            av += __shfl_xor(av, mask, 32);
        }
        l_part[m] = lv; a_part[m] = av;
    }

    if (tl15 == 0) {   // lane 0 writes rows s0..s0+7, lane 16 rows s0+8..s0+15
        const int b = b0 + wv;
#pragma unroll
        for (int m = 0; m < 8; ++m)
            y[(size_t)(s0 + rowbase + m) * BATCH + b] = a_part[m] / l_part[m];
    }
}

// ---------------------------------------------------------------------------
// Kernel 2: elementwise FFN (in-place):
//   x <- sum_j relu(x*W1[j] + b1[j]) * W2[j] + b2
// ---------------------------------------------------------------------------
__global__ __launch_bounds__(256)
void ffn_kernel(float* __restrict__ x,
                const float* __restrict__ W1, const float* __restrict__ b1,
                const float* __restrict__ W2, const float* __restrict__ b2,
                int n)
{
    __shared__ float w1s[MLP_H], b1s[MLP_H], w2s[MLP_H];
    for (int j = threadIdx.x; j < MLP_H; j += 256) {
        w1s[j] = W1[j]; b1s[j] = b1[j]; w2s[j] = W2[j];
    }
    __syncthreads();

    const int i = blockIdx.x * 256 + threadIdx.x;
    if (i < n) {
        const float x0 = x[i];
        float acc = 0.f;
#pragma unroll 4
        for (int j = 0; j < MLP_H; ++j) {
            float h = fmaf(x0, w1s[j], b1s[j]);
            h = fmaxf(h, 0.f);
            acc = fmaf(h, w2s[j], acc);
        }
        x[i] = acc + b2[0];
    }
}

// ---------------------------------------------------------------------------
// Kernel 3: stage 2, only s=0 matters for the final output.
//   x0[b] = sum_t softmax_t(q2[0,b]*k2[t,b] + bias2[0,t]) * v2[t,b]
//   out[b] = ffn2(x0[b])
// One block per b; 256 threads cooperate over t and over j.
// ---------------------------------------------------------------------------
__global__ __launch_bounds__(256)
void stage2_kernel(const float* __restrict__ yin,
                   const float* __restrict__ Wq2, const float* __restrict__ bq2,
                   const float* __restrict__ Wk2, const float* __restrict__ bk2,
                   const float* __restrict__ Wv2, const float* __restrict__ bv2,
                   const float* __restrict__ bias2,
                   const float* __restrict__ W12, const float* __restrict__ b12,
                   const float* __restrict__ W22, const float* __restrict__ b22,
                   float* __restrict__ out)
{
    const int b   = blockIdx.x;
    const int tid = threadIdx.x;

    __shared__ float ls[8], as[8];
    __shared__ float x0s;

    const float wq = Wq2[0], bqv = bq2[0];
    const float wk = Wk2[0], bkv = bk2[0];
    const float wvv = Wv2[0], bvv = bv2[0];

    const float q = fmaf(yin[b], wq, bqv);   // yin[0*BATCH + b]

    float l = 0.f, acc = 0.f;
    for (int t = tid; t < SEQ_LEN; t += 256) {
        float xv = yin[(size_t)t * BATCH + b];
        float sc = fmaf(q, fmaf(xv, wk, bkv), bias2[t]);   // bias2[0,0,t]
        float e = __expf(sc);
        l += e;
        acc = fmaf(e, fmaf(xv, wvv, bvv), acc);
    }
#pragma unroll
    for (int mask = 16; mask >= 1; mask >>= 1) {
        l   += __shfl_xor(l,   mask, 32);
        acc += __shfl_xor(acc, mask, 32);
    }
    if ((tid & 31) == 0) { ls[tid >> 5] = l; as[tid >> 5] = acc; }
    __syncthreads();
    if (tid == 0) {
        float L = 0.f, A = 0.f;
#pragma unroll
        for (int w = 0; w < 8; ++w) { L += ls[w]; A += as[w]; }
        x0s = A / L;
    }
    __syncthreads();

    const float x0 = x0s;
    float part = 0.f;
    for (int j = tid; j < MLP_H; j += 256) {
        float h = fmaf(x0, W12[j], b12[j]);
        h = fmaxf(h, 0.f);
        part = fmaf(h, W22[j], part);
    }
#pragma unroll
    for (int mask = 16; mask >= 1; mask >>= 1)
        part += __shfl_xor(part, mask, 32);
    if ((tid & 31) == 0) as[tid >> 5] = part;
    __syncthreads();
    if (tid == 0) {
        float A = 0.f;
#pragma unroll
        for (int w = 0; w < 8; ++w) A += as[w];
        out[b] = A + b22[0];
    }
}

// ---------------------------------------------------------------------------
extern "C" void kernel_launch(void* const* d_in, const int* in_sizes, int n_in,
                              void* d_out, int out_size, void* d_ws, size_t ws_size,
                              hipStream_t stream) {
    const float* src   = (const float*)d_in[0];
    const float* Wq    = (const float*)d_in[1];
    const float* bq    = (const float*)d_in[2];
    const float* Wk    = (const float*)d_in[3];
    const float* bk    = (const float*)d_in[4];
    const float* Wv    = (const float*)d_in[5];
    const float* bv    = (const float*)d_in[6];
    const float* bias1 = (const float*)d_in[7];
    const float* W1    = (const float*)d_in[8];
    const float* b1    = (const float*)d_in[9];
    const float* W2    = (const float*)d_in[10];
    const float* b2    = (const float*)d_in[11];
    const float* Wq2   = (const float*)d_in[12];
    const float* bq2   = (const float*)d_in[13];
    const float* Wk2   = (const float*)d_in[14];
    const float* bk2   = (const float*)d_in[15];
    const float* Wv2   = (const float*)d_in[16];
    const float* bv2   = (const float*)d_in[17];
    const float* bias2 = (const float*)d_in[18];
    const float* W12   = (const float*)d_in[19];
    const float* b12   = (const float*)d_in[20];
    const float* W22   = (const float*)d_in[21];
    const float* b22   = (const float*)d_in[22];
    float* outp = (float*)d_out;

    float* y = (float*)d_ws;                 // [SEQ_LEN, BATCH] f32 scratch

    // Stage 1 attention: 64 s-tiles x 16 groups of 8 b's
    attn1_kernel<<<dim3(64, 16), 256, 0, stream>>>(
        src, Wq, bq, Wk, bk, Wv, bv, bias1, y);

    // Stage 1 FFN (in-place on y)
    ffn_kernel<<<(SEQ_LEN * BATCH) / 256, 256, 0, stream>>>(
        y, W1, b1, W2, b2, SEQ_LEN * BATCH);

    // Stage 2 (s = 0 only) + FFN2 + final output
    stage2_kernel<<<BATCH, 256, 0, stream>>>(
        y, Wq2, bq2, Wk2, bk2, Wv2, bv2, bias2, W12, b12, W22, b22, outp);
}
